// PropagateLayer_30571577213074
// MI455X (gfx1250) — matrix operand order repaired and checked by reading, not compiled, not run on hardware
//
#include <hip/hip_runtime.h>

typedef __attribute__((ext_vector_type(16))) __bf16 v16bf;
typedef __attribute__((ext_vector_type(8)))  float  v8f;

#define N_PTS   50000
#define K_NB    16
#define C_CH    128
#define CVQ     16
#define E_EDGE  (N_PTS * K_NB)      // 800000 edges
#define TILES   (E_EDGE / 16)       // 50000 tiles of 16 edges
#define BN_EPS  1e-5f

// workspace layout, in floats
#define WS_MOM    0                     // 9 used (Sx,Sy,Sz,Sxx,Syy,Szz,Sxy,Sxz,Syz)
#define WS_SCALE  16                    // 128
#define WS_SHIFT  144                   // 128
#define WS_SEGSUM 512                   // N*CV
#define WS_SEGVAL (512 + N_PTS * CVQ)   // N*CV
#define WS_TOTAL  (512 + 2 * N_PTS * CVQ)

__device__ __forceinline__ float wave_reduce(float v) {
#pragma unroll
    for (int o = 16; o > 0; o >>= 1) v += __shfl_xor(v, o, 32);
    return v;
}

// ---------------- K0: zero workspace (moments + segment tables) ----------------
__global__ void k0_zero(float* __restrict__ ws) {
    int i = blockIdx.x * blockDim.x + threadIdx.x;
    if (i < WS_TOTAL) ws[i] = 0.0f;
}

// ---------------- K1: 1st/2nd moments of rel_pos (9-float reduction) ----------
// BN over p = rel @ W_pos reduces to mean/cov of rel: mu_c = Srel . w_c / E,
// E[p^2]_c = w_c^T M2 w_c / E.  Avoids 800 MB of p traffic.
__global__ void k1_moments(const float* __restrict__ rel, float* __restrict__ mom) {
    int tid    = blockIdx.x * blockDim.x + threadIdx.x;
    int stride = gridDim.x * blockDim.x;
    float sx=0.f,sy=0.f,sz=0.f,sxx=0.f,syy=0.f,szz=0.f,sxy=0.f,sxz=0.f,syz=0.f;
    for (int e = tid; e < E_EDGE; e += stride) {
        float x = rel[3*e+0], y = rel[3*e+1], z = rel[3*e+2];
        sx += x;   sy += y;   sz += z;
        sxx += x*x; syy += y*y; szz += z*z;
        sxy += x*y; sxz += x*z; syz += y*z;
    }
    sx = wave_reduce(sx);   sy = wave_reduce(sy);   sz = wave_reduce(sz);
    sxx = wave_reduce(sxx); syy = wave_reduce(syy); szz = wave_reduce(szz);
    sxy = wave_reduce(sxy); sxz = wave_reduce(sxz); syz = wave_reduce(syz);
    if ((threadIdx.x & 31) == 0) {
        atomicAdd(&mom[0], sx);  atomicAdd(&mom[1], sy);  atomicAdd(&mom[2], sz);
        atomicAdd(&mom[3], sxx); atomicAdd(&mom[4], syy); atomicAdd(&mom[5], szz);
        atomicAdd(&mom[6], sxy); atomicAdd(&mom[7], sxz); atomicAdd(&mom[8], syz);
    }
}

// ---------------- K2: fold BN into per-channel scale/shift --------------------
__global__ void k2_bnstats(const float* __restrict__ Wpos, const float* __restrict__ gamma,
                           const float* __restrict__ beta, float* __restrict__ ws) {
    int c = threadIdx.x;
    if (c >= C_CH) return;
    const float* mom = ws + WS_MOM;
    float w0 = Wpos[c], w1 = Wpos[C_CH + c], w2 = Wpos[2*C_CH + c];
    float invE = 1.0f / (float)E_EDGE;
    float mu  = (mom[0]*w0 + mom[1]*w1 + mom[2]*w2) * invE;
    float ex2 = (mom[3]*w0*w0 + mom[4]*w1*w1 + mom[5]*w2*w2
               + 2.0f*(mom[6]*w0*w1 + mom[7]*w0*w2 + mom[8]*w1*w2)) * invE;
    float var = ex2 - mu*mu;
    float sc  = gamma[c] * rsqrtf(var + BN_EPS);
    ws[WS_SCALE + c] = sc;
    ws[WS_SHIFT + c] = beta[c] - mu*sc;   // pos_emb = relu(p*scale + shift)
}

// ---------------- K3: main edge kernel (bf16 WMMA, 12 wmma / 16-edge tile) ----
__global__ __launch_bounds__(256) void k3_edges(
    const float* __restrict__ xn,   const float* __restrict__ rel,
    const float* __restrict__ Wpos,
    const float* __restrict__ Wmix, const float* __restrict__ bmix,
    const float* __restrict__ Wval, const float* __restrict__ bval,
    const int*   __restrict__ nidx, float* __restrict__ ws)
{
    // B fragments pre-swizzled to WMMA layout: lane l holds column N=l&15,
// K = 32b + 16*(i>>3) + 8*(l>>4) + (i&7)  (i = element 0..15)  ->  one 32B run per lane.
    __shared__ __align__(32) __bf16 lds_bm[8][32][16];   // W_mix  (K=256)
    __shared__ __align__(32) __bf16 lds_bv[4][32][16];   // W_val  (K=128)
    __shared__ float4 lds_pc[C_CH];                      // {w0,w1,w2,scale}
    __shared__ float  lds_sh[C_CH];                      // shift

    const float* scale  = ws + WS_SCALE;
    const float* shift  = ws + WS_SHIFT;
    float* segsum = ws + WS_SEGSUM;
    float* segval = ws + WS_SEGVAL;

    int t = threadIdx.x;
    for (int j = t; j < 8*32*16; j += 256) {
        int b = j >> 9, l = (j >> 4) & 31, i = j & 15;
        int k = b*32 + ((i >> 3) << 4) + ((l >> 4) << 3) + (i & 7);
        lds_bm[b][l][i] = (__bf16)Wmix[k*CVQ + (l & 15)];
    }
    for (int j = t; j < 4*32*16; j += 256) {
        int b = j >> 9, l = (j >> 4) & 31, i = j & 15;
        int k = b*32 + ((i >> 3) << 4) + ((l >> 4) << 3) + (i & 7);
        lds_bv[b][l][i] = (__bf16)Wval[k*CVQ + (l & 15)];
    }
    if (t < C_CH) {
        lds_pc[t] = make_float4(Wpos[t], Wpos[C_CH + t], Wpos[2*C_CH + t], scale[t]);
        lds_sh[t] = shift[t];
    }
    __syncthreads();

    int lane = t & 31;
    int m    = lane & 15;   // A row / C column (N)
    int kh   = lane >> 4;   // K-half selector
    float bm_bias = bmix[m];
    float bv_bias = bval[m];

    // Keep the 12 B fragments register-resident (96 VGPRs, feed WMMA directly).
    v16bf Bm0 = *(const v16bf*)&lds_bm[0][lane][0];
    v16bf Bm1 = *(const v16bf*)&lds_bm[1][lane][0];
    v16bf Bm2 = *(const v16bf*)&lds_bm[2][lane][0];
    v16bf Bm3 = *(const v16bf*)&lds_bm[3][lane][0];
    v16bf Bm4 = *(const v16bf*)&lds_bm[4][lane][0];
    v16bf Bm5 = *(const v16bf*)&lds_bm[5][lane][0];
    v16bf Bm6 = *(const v16bf*)&lds_bm[6][lane][0];
    v16bf Bm7 = *(const v16bf*)&lds_bm[7][lane][0];
    v16bf Bv0 = *(const v16bf*)&lds_bv[0][lane][0];
    v16bf Bv1 = *(const v16bf*)&lds_bv[1][lane][0];
    v16bf Bv2 = *(const v16bf*)&lds_bv[2][lane][0];
    v16bf Bv3 = *(const v16bf*)&lds_bv[3][lane][0];
    v16bf BmLo[4] = {Bm0, Bm1, Bm2, Bm3};
    v16bf BmHi[4] = {Bm4, Bm5, Bm6, Bm7};
    v16bf Bv[4]   = {Bv0, Bv1, Bv2, Bv3};

    int wave = blockIdx.x * (blockDim.x >> 5) + (t >> 5);
    int nwav = gridDim.x  * (blockDim.x >> 5);

    for (int tile = wave; tile < TILES; tile += nwav) {   // wave-uniform: EXEC all-1s
        // Don't let loop-invariant LDS reads (pos constants) get hoisted into
        // ~320 VGPRs: reload per tile (broadcast DS reads), keep regs < 256.
        asm volatile("" ::: "memory");

        int e0 = tile * 16;

        // prefetch next tile's xn block: 16 rows x 512 B = 8 KB, 256 B per lane
        int tnext = tile + nwav;
        if (tnext < TILES) {
            const float* pf = xn + (size_t)tnext * 16 * C_CH + (size_t)lane * 64;
            __builtin_prefetch(pf, 0, 0);   // global_prefetch_b8
        }

        const float* rp = rel + (size_t)(e0 + m) * 3;
        float rx = rp[0], ry = rp[1], rz = rp[2];

        v8f cm, cvacc;
#pragma unroll
        for (int v = 0; v < 8; v++) { cm[v] = bm_bias; cvacc[v] = bv_bias; }

        // mixer K = 0..127 : pos_emb computed straight into A-fragment layout
#pragma unroll
        for (int b = 0; b < 4; b++) {
            int cb = b*32 + kh*8;
            v16bf a;
#pragma unroll
            for (int i = 0; i < 8; i++) {
                float4 pc = lds_pc[cb + i];
                float p  = fmaf(rx, pc.x, fmaf(ry, pc.y, rz * pc.z));
                float pe = fmaf(p, pc.w, lds_sh[cb + i]);
                a[i] = (__bf16)(pe > 0.f ? pe : 0.f);
                float4 pc2 = lds_pc[cb + 16 + i];
                float p2  = fmaf(rx, pc2.x, fmaf(ry, pc2.y, rz * pc2.z));
                float pe2 = fmaf(p2, pc2.w, lds_sh[cb + 16 + i]);
                a[i + 8] = (__bf16)(pe2 > 0.f ? pe2 : 0.f);
            }
            cm = __builtin_amdgcn_wmma_f32_16x16x32_bf16(false, a, false, BmLo[b],
                                                         (short)0, cm, false, false);
        }

        // mixer K = 128..255 and values K = 0..127 share the xn A-fragments
        const float* xrow = xn + (size_t)(e0 + m) * C_CH;
#pragma unroll
        for (int b = 0; b < 4; b++) {
            int cb = b*32 + kh*8;
            float4 f0 = *(const float4*)(xrow + cb);
            float4 f1 = *(const float4*)(xrow + cb + 4);
            float4 f2 = *(const float4*)(xrow + cb + 16);
            float4 f3 = *(const float4*)(xrow + cb + 20);
            v16bf a;
            a[0]=(__bf16)f0.x;  a[1]=(__bf16)f0.y;  a[2]=(__bf16)f0.z;  a[3]=(__bf16)f0.w;
            a[4]=(__bf16)f1.x;  a[5]=(__bf16)f1.y;  a[6]=(__bf16)f1.z;  a[7]=(__bf16)f1.w;
            a[8]=(__bf16)f2.x;  a[9]=(__bf16)f2.y;  a[10]=(__bf16)f2.z; a[11]=(__bf16)f2.w;
            a[12]=(__bf16)f3.x; a[13]=(__bf16)f3.y; a[14]=(__bf16)f3.z; a[15]=(__bf16)f3.w;
            cm    = __builtin_amdgcn_wmma_f32_16x16x32_bf16(false, a, false, BmHi[b],
                                                            (short)0, cm, false, false);
            cvacc = __builtin_amdgcn_wmma_f32_16x16x32_bf16(false, a, false, Bv[b],
                                                            (short)0, cvacc, false, false);
        }

        // softmax numerators (shift-invariant: relu output is bounded, no max pass)
        // C-fragment: VGPR v, lane l -> row M = v + 8*kh (edge), col N = m (channel)
#pragma unroll
        for (int v = 0; v < 8; v++) {
            float mo = cm[v];    mo = mo > 0.f ? mo : 0.f;
            float ex = __expf(mo);
            float va = cvacc[v]; va = va > 0.f ? va : 0.f;
            int seg = nidx[e0 + kh*8 + v];
            int col = seg * CVQ + m;
            atomicAdd(&segsum[col], ex);
            atomicAdd(&segval[col], va * ex);
        }
    }
}

// ---------------- K4: normalize, tile x8, add residual ------------------------
__global__ void k4_final(const float* __restrict__ xc, const float* __restrict__ ws,
                         float* __restrict__ out) {
    int i = blockIdx.x * blockDim.x + threadIdx.x;
    if (i >= N_PTS * C_CH) return;
    int n = i >> 7, c = i & (CVQ - 1);           // tiled: out col c uses agg[c % 16]
    float s = ws[WS_SEGSUM + n*CVQ + c];
    float v = ws[WS_SEGVAL + n*CVQ + c];
    float agg = (s > 0.f) ? (v / s) : 0.f;       // empty segments contribute 0
    out[i] = xc[i] + agg;
}

extern "C" void kernel_launch(void* const* d_in, const int* in_sizes, int n_in,
                              void* d_out, int out_size, void* d_ws, size_t ws_size,
                              hipStream_t stream) {
    const float* x_center = (const float*)d_in[0];
    const float* x_nb     = (const float*)d_in[1];
    const float* rel      = (const float*)d_in[2];
    const float* Wpos     = (const float*)d_in[3];
    const float* gamma    = (const float*)d_in[4];
    const float* beta     = (const float*)d_in[5];
    const float* Wmix     = (const float*)d_in[6];
    const float* bmix     = (const float*)d_in[7];
    const float* Wval     = (const float*)d_in[8];
    const float* bval     = (const float*)d_in[9];
    const int*   nidx     = (const int*)d_in[10];
    float* ws  = (float*)d_ws;
    float* out = (float*)d_out;

    k0_zero   <<<(WS_TOTAL + 255) / 256, 256, 0, stream>>>(ws);
    k1_moments<<<512, 256, 0, stream>>>(rel, ws + WS_MOM);
    k2_bnstats<<<1, 128, 0, stream>>>(Wpos, gamma, beta, ws);
    k3_edges  <<<512, 256, 0, stream>>>(x_nb, rel, Wpos, Wmix, bmix, Wval, bval, nidx, ws);
    k4_final  <<<(N_PTS * C_CH + 255) / 256, 256, 0, stream>>>(x_center, ws, out);
}
